// LSTMDecoder_3977139716498
// MI455X (gfx1250) — matrix operand order repaired and checked by reading, not compiled, not run on hardware
//
#include <hip/hip_runtime.h>
#include <hip/hip_bf16.h>
#include <stdint.h>

// ---------------------------------------------------------------------------
// Problem constants (from reference)
// ---------------------------------------------------------------------------
#define VSZ 32000
#define ESZ 512
#define HSZ 1024
#define LSZ 2
#define BSZ 32
#define TSZ 64
#define SSZ 64
#define G4H (4 * HSZ)          // 4096
#define KX  (ESZ + HSZ)        // 1536

// ---------------------------------------------------------------------------
// WMMA vector types (gfx1250, wave32)
// ---------------------------------------------------------------------------
typedef __attribute__((ext_vector_type(16))) __bf16 v16bf;
typedef __attribute__((ext_vector_type(8)))  __bf16 v8bf;
typedef __attribute__((ext_vector_type(8)))  float  v8f;

__device__ __forceinline__ unsigned short f2bf(float f) {
    unsigned int u = __float_as_uint(f);
    unsigned int r = u + 0x7FFFu + ((u >> 16) & 1u);   // round-to-nearest-even
    return (unsigned short)(r >> 16);
}
__device__ __forceinline__ float sigmoidf_(float x) {
    return 1.0f / (1.0f + expf(-x));
}

// A-fragment (16x32 bf16): lane holds row (lane&15), K chunks [klo,klo+8) and
// [klo+16,klo+24), klo=(lane>>4)*8 -> two contiguous 16B loads.
__device__ __forceinline__ v16bf load_a_frag(const unsigned short* p) {
    v8bf lo = *reinterpret_cast<const v8bf*>(p);
    v8bf hi = *reinterpret_cast<const v8bf*>(p + 16);
    return __builtin_shufflevector(lo, hi, 0,1,2,3,4,5,6,7,8,9,10,11,12,13,14,15);
}
// B-fragment (32x16 bf16): lane holds column (lane&15), K chunks [klo,klo+8) and
// [klo+16,klo+24), klo=(lane>>4)*8 -> two contiguous 16B loads from row-major W[N,K].
__device__ __forceinline__ v16bf load_b_frag(const unsigned short* p) {
    v8bf lo = *reinterpret_cast<const v8bf*>(p);
    v8bf hi = *reinterpret_cast<const v8bf*>(p + 16);
    return __builtin_shufflevector(lo, hi, 0,1,2,3,4,5,6,7,8,9,10,11,12,13,14,15);
}
__device__ __forceinline__ v8f wmma_bf16(v16bf a, v16bf b, v8f c) {
    return __builtin_amdgcn_wmma_f32_16x16x32_bf16(
        false, a, false, b, (short)0, c, false, false);
}

// ---------------------------------------------------------------------------
// bf16 WMMA GEMM, wave = full 32(M) x 16(N) stripe (two accumulators, B-frag
// reused for both M-tiles). Block = 128 threads = 4 waves -> 32 x 64 tile.
// Software-pipelined K loop (double-buffered fragments, peeled epilogue so
// the steady-state body is branch-free).
// blockIdx.z selects GEMM pair 0 (A1*W1^T + bias1 -> D1) or 1 (A2... -> D2).
// Grid: (N/64, M/32, npairs).
// ---------------------------------------------------------------------------
__global__ __launch_bounds__(128)
void wmma_gemm_kernel(const unsigned short* __restrict__ A1, int lda1,
                      const unsigned short* __restrict__ W1, int ldw1, int K1,
                      const float* __restrict__ bias1,
                      float* __restrict__ D1,
                      const unsigned short* __restrict__ A2, int lda2,
                      const unsigned short* __restrict__ W2, int ldw2, int K2,
                      const float* __restrict__ bias2,
                      float* __restrict__ D2,
                      int ldd) {
    const int lane = threadIdx.x & 31;
    const int wave = threadIdx.x >> 5;
    const int mbase = blockIdx.y * 32;
    const int nbase = blockIdx.x * 64 + wave * 16;

    const unsigned short* A = A1;
    const unsigned short* W = W1;
    const float* bias = bias1;
    float* D = D1;
    int lda = lda1, ldw = ldw1, K = K1;
    if (blockIdx.z == 1) {
        A = A2; W = W2; bias = bias2; D = D2; lda = lda2; ldw = ldw2; K = K2;
    }

    const int aklo = (lane >> 4) << 3;          // 0 or 8
    const unsigned short* aptr0 = A + (long)(mbase + (lane & 15)) * lda + aklo;
    const unsigned short* aptr1 = aptr0 + (long)16 * lda;   // rows mbase+16..31
    const int ncol = nbase + (lane & 15);
    const unsigned short* wptr = W + (long)ncol * ldw + ((lane >> 4) << 3);

    v8f acc0 = {0.f, 0.f, 0.f, 0.f, 0.f, 0.f, 0.f, 0.f};
    v8f acc1 = acc0;

    // Software pipeline: loads for step k0 fly while WMMA consumes k0-32.
    v16bf a0 = load_a_frag(aptr0);
    v16bf a1 = load_a_frag(aptr1);
    v16bf b  = load_b_frag(wptr);

    // Branch-free steady state (final iteration peeled below).
#pragma unroll 2
    for (int k0 = 32; k0 < K; k0 += 32) {
        __builtin_prefetch(wptr + k0 + 96, 0, 3);   // speculative; safe past end
        v16bf a0n = load_a_frag(aptr0 + k0);
        v16bf a1n = load_a_frag(aptr1 + k0);
        v16bf bn  = load_b_frag(wptr + k0);
        acc0 = wmma_bf16(a0, b, acc0);   // B fragment reused for both M tiles
        acc1 = wmma_bf16(a1, b, acc1);
        a0 = a0n; a1 = a1n; b = bn;
    }
    acc0 = wmma_bf16(a0, b, acc0);
    acc1 = wmma_bf16(a1, b, acc1);

    const float bsum = bias ? bias[ncol] : 0.f;
    const int r0 = mbase + ((lane >> 4) << 3);          // +0 or +8
#pragma unroll
    for (int r = 0; r < 8; ++r) {
        D[(long)(r0 + r) * ldd + ncol]      = acc0[r] + bsum;
        D[(long)(r0 + 16 + r) * ldd + ncol] = acc1[r] + bsum;
    }
}

// ---------------------------------------------------------------------------
// Utility kernels
// ---------------------------------------------------------------------------
__global__ void convert_f32_bf16_kernel(const float* __restrict__ in,
                                        unsigned short* __restrict__ out, int n) {
    int i = blockIdx.x * 256 + threadIdx.x;
    if (i < n) out[i] = f2bf(in[i]);
}

// embedded_bf16 layout [T, B, E]
__global__ void embed_kernel(const int* __restrict__ tgt,
                             const float* __restrict__ emb,
                             unsigned short* __restrict__ out) {
    int i = blockIdx.x * 256 + threadIdx.x;
    if (i >= TSZ * BSZ * ESZ) return;
    int e = i & (ESZ - 1);
    int tb = i >> 9;              // E = 512
    int b = tb & (BSZ - 1);
    int t = tb >> 5;              // B = 32
    int tok = tgt[b * TSZ + t];
    out[i] = f2bf(emb[(long)tok * ESZ + e]);
}

// Fused attention: scores -> softmax -> ctx; assembles x_step = [emb_t | ctx].
// One block per batch element, 256 threads (8 waves).
__global__ __launch_bounds__(256)
void attn_kernel(const float* __restrict__ q,          // [B,H] = dec_h@W1^T + b1
                 const float* __restrict__ enc_proj,   // [B,S,H]
                 const float* __restrict__ enc_out,    // [B,S,H] fp32
                 const float* __restrict__ v,
                 const float* __restrict__ bv,
                 const unsigned char* __restrict__ mask,
                 const unsigned short* __restrict__ emb_t,  // [B,E] bf16 slice
                 unsigned short* __restrict__ x_step) {     // [B, E+H] bf16
    __shared__ float q_sh[HSZ];
    __shared__ float sc[SSZ];
    const int b = blockIdx.x;
    const int tid = threadIdx.x;
    const int lane = tid & 31;
    const int wave = tid >> 5;

    for (int i = tid; i < HSZ; i += 256) q_sh[i] = q[b * HSZ + i];
    for (int i = tid; i < ESZ; i += 256)
        x_step[(long)b * KX + i] = emb_t[b * ESZ + i];
    __syncthreads();

    for (int s = wave; s < SSZ; s += 8) {
        const float* ep = enc_proj + ((long)b * SSZ + s) * HSZ;
        float p = 0.f;
        for (int h = lane; h < HSZ; h += 32)
            p += v[h] * tanhf(q_sh[h] + ep[h]);
#pragma unroll
        for (int off = 16; off > 0; off >>= 1)
            p += __shfl_xor(p, off, 32);
        if (lane == 0) {
            float sval = p + bv[0];
            if (!mask[b * SSZ + s]) sval = -1e9f;
            sc[s] = sval;
        }
    }
    __syncthreads();

    if (tid == 0) {
        float m = sc[0];
        for (int s = 1; s < SSZ; ++s) m = fmaxf(m, sc[s]);
        float sum = 0.f;
        for (int s = 0; s < SSZ; ++s) { float e = expf(sc[s] - m); sc[s] = e; sum += e; }
        float inv = 1.f / sum;
        for (int s = 0; s < SSZ; ++s) sc[s] *= inv;
    }
    __syncthreads();

    for (int h = tid; h < HSZ; h += 256) {
        float acc = 0.f;
        const float* eo = enc_out + (long)b * SSZ * HSZ + h;
#pragma unroll 4
        for (int s = 0; s < SSZ; ++s) acc += sc[s] * eo[(long)s * HSZ];
        x_step[(long)b * KX + ESZ + h] = f2bf(acc);
    }
}

// LSTM cell: gates = gA + gB (two partial GEMM outputs), order i,f,g,o.
__global__ void lstm_cell_kernel(const float* __restrict__ gA,
                                 const float* __restrict__ gB,
                                 float* __restrict__ c,
                                 float* __restrict__ h,
                                 unsigned short* __restrict__ h_bf,
                                 float* __restrict__ dec_out) { // pre-offset by t*H, or null
    int idx = blockIdx.x * 256 + threadIdx.x;
    if (idx >= BSZ * HSZ) return;
    int b = idx >> 10;             // H = 1024
    int hh = idx & (HSZ - 1);
    const float* ga = gA + (long)b * G4H;
    const float* gb = gB + (long)b * G4H;
    float ig = sigmoidf_(ga[hh]            + gb[hh]);
    float fg = sigmoidf_(ga[HSZ + hh]      + gb[HSZ + hh]);
    float gg = tanhf    (ga[2 * HSZ + hh]  + gb[2 * HSZ + hh]);
    float og = sigmoidf_(ga[3 * HSZ + hh]  + gb[3 * HSZ + hh]);
    float cn = fg * c[idx] + ig * gg;
    float hn = og * tanhf(cn);
    c[idx] = cn;
    h[idx] = hn;
    h_bf[idx] = f2bf(hn);
    if (dec_out) dec_out[(long)b * (TSZ * HSZ) + hh] = hn;
}

__global__ void finalize_kernel(const float* __restrict__ h,
                                const float* __restrict__ c,
                                float* __restrict__ out_h,
                                float* __restrict__ out_c) {
    int i = blockIdx.x * 256 + threadIdx.x;
    if (i < LSZ * BSZ * HSZ) { out_h[i] = h[i]; out_c[i] = c[i]; }
}

// ---------------------------------------------------------------------------
// Host: kernel_launch
// ---------------------------------------------------------------------------
static inline size_t align256(size_t x) { return (x + 255) & ~(size_t)255; }

extern "C" void kernel_launch(void* const* d_in, const int* in_sizes, int n_in,
                              void* d_out, int out_size, void* d_ws, size_t ws_size,
                              hipStream_t stream) {
    (void)in_sizes; (void)n_in; (void)out_size; (void)ws_size;

    const int*            tgt   = (const int*)           d_in[0];
    const float*          enc   = (const float*)         d_in[1];
    const float*          h0    = (const float*)         d_in[2];
    const float*          c0    = (const float*)         d_in[3];
    const unsigned char*  mask  = (const unsigned char*) d_in[4];
    const float*          emb   = (const float*)         d_in[5];
    const float*          W1    = (const float*)         d_in[6];
    const float*          b1    = (const float*)         d_in[7];
    const float*          W2    = (const float*)         d_in[8];
    const float*          b2    = (const float*)         d_in[9];
    const float*          vv    = (const float*)         d_in[10];
    const float*          bv    = (const float*)         d_in[11];
    const float*          W_ih0 = (const float*)         d_in[12];
    const float*          W_hh0 = (const float*)         d_in[13];
    const float*          b_ih0 = (const float*)         d_in[14];
    const float*          b_hh0 = (const float*)         d_in[15];
    const float*          W_ih1 = (const float*)         d_in[16];
    const float*          W_hh1 = (const float*)         d_in[17];
    const float*          b_ih1 = (const float*)         d_in[18];
    const float*          b_hh1 = (const float*)         d_in[19];

    float* out = (float*)d_out;   // [dec_out(B,T,H) | h_n(L,B,H) | c_n(L,B,H)]

    // ---- workspace carve-up ------------------------------------------------
    char* ws = (char*)d_ws;
    size_t cur = 0;
    auto take = [&](size_t bytes) { void* p = ws + cur; cur = align256(cur + bytes); return p; };

    unsigned short* embedded = (unsigned short*)take((size_t)TSZ * BSZ * ESZ * 2);   // [T,B,E]
    unsigned short* enc_bf   = (unsigned short*)take((size_t)BSZ * SSZ * HSZ * 2);
    float*          enc_proj = (float*)         take((size_t)BSZ * SSZ * HSZ * 4);
    unsigned short* W1_bf    = (unsigned short*)take((size_t)HSZ * HSZ * 2);
    unsigned short* W2_bf    = (unsigned short*)take((size_t)HSZ * HSZ * 2);
    unsigned short* Wih0_bf  = (unsigned short*)take((size_t)G4H * KX  * 2);
    unsigned short* Whh0_bf  = (unsigned short*)take((size_t)G4H * HSZ * 2);
    unsigned short* Wih1_bf  = (unsigned short*)take((size_t)G4H * HSZ * 2);
    unsigned short* Whh1_bf  = (unsigned short*)take((size_t)G4H * HSZ * 2);
    float*          qbuf     = (float*)         take((size_t)BSZ * HSZ * 4);
    unsigned short* x_step   = (unsigned short*)take((size_t)BSZ * KX * 2);
    float*          gatesA   = (float*)         take((size_t)BSZ * G4H * 4);
    float*          gatesB   = (float*)         take((size_t)BSZ * G4H * 4);
    float*          h_st     = (float*)         take((size_t)LSZ * BSZ * HSZ * 4);
    float*          c_st     = (float*)         take((size_t)LSZ * BSZ * HSZ * 4);
    unsigned short* h_bf     = (unsigned short*)take((size_t)LSZ * BSZ * HSZ * 2);

    // ---- one-time setup (re-done each call for determinism) ---------------
    auto cvt = [&](const float* src, unsigned short* dst, int n) {
        convert_f32_bf16_kernel<<<(n + 255) / 256, 256, 0, stream>>>(src, dst, n);
    };
    embed_kernel<<<(TSZ * BSZ * ESZ + 255) / 256, 256, 0, stream>>>(tgt, emb, embedded);
    cvt(enc,   enc_bf,  BSZ * SSZ * HSZ);
    cvt(W1,    W1_bf,   HSZ * HSZ);
    cvt(W2,    W2_bf,   HSZ * HSZ);
    cvt(W_ih0, Wih0_bf, G4H * KX);
    cvt(W_hh0, Whh0_bf, G4H * HSZ);
    cvt(W_ih1, Wih1_bf, G4H * HSZ);
    cvt(W_hh1, Whh1_bf, G4H * HSZ);

    hipMemcpyAsync(h_st, h0, (size_t)LSZ * BSZ * HSZ * 4, hipMemcpyDeviceToDevice, stream);
    hipMemcpyAsync(c_st, c0, (size_t)LSZ * BSZ * HSZ * 4, hipMemcpyDeviceToDevice, stream);
    cvt(h0, h_bf, LSZ * BSZ * HSZ);

    // enc_proj = enc @ W2^T + b2  :  M = B*S = 2048, N = H, K = H
    {
        dim3 grid(HSZ / 64, (BSZ * SSZ) / 32, 1);
        wmma_gemm_kernel<<<grid, 128, 0, stream>>>(
            enc_bf, HSZ, W2_bf, HSZ, HSZ, b2, enc_proj,
            nullptr, 0, nullptr, 0, 0, nullptr, nullptr, HSZ);
    }

    unsigned short* h_bf_l0 = h_bf;
    unsigned short* h_bf_l1 = h_bf + BSZ * HSZ;
    float* c_l0 = c_st;
    float* c_l1 = c_st + BSZ * HSZ;
    float* h_l0 = h_st;
    float* h_l1 = h_st + BSZ * HSZ;

    // ---- 64 sequential decode steps ---------------------------------------
    for (int t = 0; t < TSZ; ++t) {
        // q = dec_h @ W1^T + b1   (M=32, N=H, K=H)
        {
            dim3 grid(HSZ / 64, 1, 1);
            wmma_gemm_kernel<<<grid, 128, 0, stream>>>(
                h_bf_l1, HSZ, W1_bf, HSZ, HSZ, b1, qbuf,
                nullptr, 0, nullptr, 0, 0, nullptr, nullptr, HSZ);
        }
        // fused attention -> x_step = [emb_t | ctx]
        attn_kernel<<<BSZ, 256, 0, stream>>>(
            qbuf, enc_proj, enc, vv, bv, mask,
            embedded + (size_t)t * BSZ * ESZ, x_step);

        // layer 0 gates: z=0: x@W_ih0^T + b_ih0 -> gatesA
        //                z=1: h0@W_hh0^T + b_hh0 -> gatesB
        {
            dim3 grid(G4H / 64, 1, 2);
            wmma_gemm_kernel<<<grid, 128, 0, stream>>>(
                x_step, KX, Wih0_bf, KX, KX, b_ih0, gatesA,
                h_bf_l0, HSZ, Whh0_bf, HSZ, HSZ, b_hh0, gatesB, G4H);
        }
        lstm_cell_kernel<<<(BSZ * HSZ + 255) / 256, 256, 0, stream>>>(
            gatesA, gatesB, c_l0, h_l0, h_bf_l0, nullptr);

        // layer 1 gates
        {
            dim3 grid(G4H / 64, 1, 2);
            wmma_gemm_kernel<<<grid, 128, 0, stream>>>(
                h_bf_l0, HSZ, Wih1_bf, HSZ, HSZ, b_ih1, gatesA,
                h_bf_l1, HSZ, Whh1_bf, HSZ, HSZ, b_hh1, gatesB, G4H);
        }
        lstm_cell_kernel<<<(BSZ * HSZ + 255) / 256, 256, 0, stream>>>(
            gatesA, gatesB, c_l1, h_l1, h_bf_l1, out + (size_t)t * HSZ);
    }

    // ---- h_n, c_n ----------------------------------------------------------
    finalize_kernel<<<(LSZ * BSZ * HSZ + 255) / 256, 256, 0, stream>>>(
        h_st, c_st,
        out + (size_t)BSZ * TSZ * HSZ,
        out + (size_t)BSZ * TSZ * HSZ + (size_t)LSZ * BSZ * HSZ);
}
